// GAT_22454089023552
// MI455X (gfx1250) — matrix operand order repaired and checked by reading, not compiled, not run on hardware
//
#include <hip/hip_runtime.h>
#include <hip/hip_bf16.h>
#include <stdint.h>

// ---------------------------------------------------------------------------
// Types for CDNA5 WMMA (wave32): V_WMMA_F32_16X16X32_BF16
// ---------------------------------------------------------------------------
typedef __attribute__((ext_vector_type(16))) __bf16        v16bf;
typedef __attribute__((ext_vector_type(8)))  float         v8f;
typedef __attribute__((ext_vector_type(8)))  unsigned int  u32x8;
typedef __attribute__((ext_vector_type(4)))  unsigned int  u32x4;

#define SLOPE 0.2f

static __device__ __forceinline__ unsigned short f2bf(float f) {
  unsigned u = __float_as_uint(f);
  u += 0x7FFFu + ((u >> 16) & 1u);   // round-to-nearest-even
  return (unsigned short)(u >> 16);
}
static __device__ __forceinline__ unsigned fkey(float f) {
  unsigned u = __float_as_uint(f);
  return (u & 0x80000000u) ? ~u : (u | 0x80000000u);  // order-preserving key, all real keys > 0
}
static __device__ __forceinline__ float fdecode(unsigned k) {
  unsigned u = (k & 0x80000000u) ? (k & 0x7FFFFFFFu) : ~k;
  return __uint_as_float(u);
}
static __device__ __forceinline__ float lrelu(float x) { return x >= 0.f ? x : SLOPE * x; }

// ---------------------------------------------------------------------------
// float -> bf16 conversion, optional ReLU, zero pad columns [valid, dstride)
// ---------------------------------------------------------------------------
__global__ void cvt_bf16_kernel(const float* __restrict__ src, int sstride, int valid,
                                unsigned short* __restrict__ dst, int dstride,
                                long total, int relu) {
  long i = (long)blockIdx.x * blockDim.x + threadIdx.x;
  if (i >= total) return;
  int r = (int)(i / dstride), c = (int)(i % dstride);
  float v = 0.f;
  if (c < valid) {
    v = src[(long)r * sstride + c];
    if (relu) v = fmaxf(v, 0.f);
  }
  dst[i] = f2bf(v);
}

// W (Ksrc x Hsrc, row-major) -> Wt (Np x Kp, N-major bf16, zero-padded)
__global__ void conv_w_kernel(const float* __restrict__ W, unsigned short* __restrict__ Wt,
                              int Ksrc, int Hsrc, int Kp, int Np) {
  int i = blockIdx.x * blockDim.x + threadIdx.x;
  if (i >= Np * Kp) return;
  int n = i / Kp, k = i % Kp;
  float v = (n < Hsrc && k < Ksrc) ? W[(long)k * Hsrc + n] : 0.f;
  Wt[i] = f2bf(v);
}

// ---------------------------------------------------------------------------
// WMMA GEMM: C[M x CS] (f32) = A[M x Kp] (bf16 row-major) * Bt[Np x Kp]^T
// (Bt is N-major / pre-transposed, zero-padded). One 16x16 tile per wave.
// Register fragments per ISA 7.12.2 bf16 layouts:
//   A: lane l (m=l&15, khalf=l>>4): elems 0..7 = K kb..kb+7, 8..15 = kb+16..kb+23,
//      kb = kstep + khalf*8
//   B: lane l (n=l&15): same K order as A (elems 0..7 = kb..kb+7, 8..15 = kb+16..kb+23)
// ---------------------------------------------------------------------------
__global__ void __launch_bounds__(256)
gemm_bf16_wmma(const unsigned short* __restrict__ A,
               const unsigned short* __restrict__ Bt,
               float* __restrict__ C,
               int M, int Kp, int Np, int CS) {
  const int lane = threadIdx.x & 31;
  const int wave = threadIdx.x >> 5;
  const int tm = blockIdx.x;
  const int tn = blockIdx.y * 8 + wave;
  if (tn * 16 >= Np) return;                 // wave-uniform: EXEC stays all-ones
  const int mrow = tm * 16 + (lane & 15);
  const int nrow = tn * 16 + (lane & 15);
  const int khalf = lane >> 4;
  const size_t arow = (size_t)(mrow < M ? mrow : M - 1) * Kp;
  const size_t brow = (size_t)nrow * Kp;

  v8f acc = {};
  for (int k = 0; k < Kp; k += 32) {
    const int ka = k + khalf * 8;
    u32x4 q0 = *(const u32x4*)(A + arow + ka);        // K ka..ka+7
    u32x4 q1 = *(const u32x4*)(A + arow + ka + 16);   // K ka+16..ka+23
    u32x8 ar;
    ar[0] = q0[0]; ar[1] = q0[1]; ar[2] = q0[2]; ar[3] = q0[3];
    ar[4] = q1[0]; ar[5] = q1[1]; ar[6] = q1[2]; ar[7] = q1[3];
    const int kb = k + khalf * 8;
    u32x4 p0 = *(const u32x4*)(Bt + brow + kb);       // K kb..kb+7
    u32x4 p1 = *(const u32x4*)(Bt + brow + kb + 16);  // K kb+16..kb+23
    u32x8 br;
    br[0] = p0[0]; br[1] = p0[1]; br[2] = p0[2]; br[3] = p0[3];
    br[4] = p1[0]; br[5] = p1[1]; br[6] = p1[2]; br[7] = p1[3];
    v16bf av = __builtin_bit_cast(v16bf, ar);
    v16bf bv = __builtin_bit_cast(v16bf, br);
    acc = __builtin_amdgcn_wmma_f32_16x16x32_bf16(
        false, av, false, bv, (short)0, acc, false, false);
  }
  // C/D layout: VGPR r -> row tm*16 + r + (lane>=16 ? 8 : 0), col = tn*16 + (lane&15)
  const int col = tn * 16 + (lane & 15);
  const int rbase = tm * 16 + khalf * 8;
#pragma unroll
  for (int r = 0; r < 8; ++r) {
    int row = rbase + r;
    if (row < M) C[(size_t)row * CS + col] = acc[r];
  }
}

// ---------------------------------------------------------------------------
// Per-node attention scalars: oA[n] = hA[n,:]·aA ; oB[n] = hB[n,:]·aB  (wave/node)
// ---------------------------------------------------------------------------
__global__ void node_dots_kernel(const float* __restrict__ hA, const float* __restrict__ aA,
                                 const float* __restrict__ hB, const float* __restrict__ aB,
                                 float* __restrict__ oA, float* __restrict__ oB,
                                 int Nn, int H, int HS) {
  int w = (blockIdx.x * blockDim.x + threadIdx.x) >> 5;
  int lane = threadIdx.x & 31;
  if (w >= Nn) return;
  float sA = 0.f, sB = 0.f;
  for (int c = lane; c < H; c += 32) {
    sA += hA[(size_t)w * HS + c] * aA[c];
    sB += hB[(size_t)w * HS + c] * aB[c];
  }
#pragma unroll
  for (int o = 16; o > 0; o >>= 1) {
    sA += __shfl_down(sA, o, 32);
    sB += __shfl_down(sB, o, 32);
  }
  if (lane == 0) { oA[w] = sA; oB[w] = sB; }
}

__global__ void init_nodes_kernel(unsigned* __restrict__ mu, float* __restrict__ denom, int Nn) {
  int i = blockIdx.x * blockDim.x + threadIdx.x;
  if (i >= Nn) return;
  mu[i] = 0u;          // key of "smaller than any real float"
  denom[i] = 0.f;
}

__global__ void init_bias_kernel(float* __restrict__ out, const float* __restrict__ b,
                                 long total, int H, int stride) {
  long i = (long)blockIdx.x * blockDim.x + threadIdx.x;
  if (i >= total) return;
  int c = (int)(i % stride);
  out[i] = (c < H) ? b[c] : 0.f;
}

__global__ void decode_m_kernel(unsigned* __restrict__ mu, int Nn) {
  int i = blockIdx.x * blockDim.x + threadIdx.x;
  if (i >= Nn) return;
  float f = fdecode(mu[i]);
  ((float*)mu)[i] = f;   // in-place: buffer becomes mf
}

// ---------------------------------------------------------------------------
// Edge passes. Edges e < E come from edge_index; e >= E are self loops.
// ---------------------------------------------------------------------------
__global__ void edge_max_kernel(const int* __restrict__ src, const int* __restrict__ dst,
                                int E, int Nn,
                                const float* __restrict__ as_, const float* __restrict__ ad_,
                                unsigned* __restrict__ mu) {
  int e = blockIdx.x * blockDim.x + threadIdx.x;
  if (e >= E + Nn) return;
  int s, d;
  if (e < E) { s = src[e]; d = dst[e]; } else { s = d = e - E; }
  float v = lrelu(as_[s] + ad_[d]);
  __hip_atomic_fetch_max(&mu[d], fkey(v), __ATOMIC_RELAXED, __HIP_MEMORY_SCOPE_AGENT);
}

__global__ void edge_denom_kernel(const int* __restrict__ src, const int* __restrict__ dst,
                                  int E, int Nn,
                                  const float* __restrict__ as_, const float* __restrict__ ad_,
                                  const float* __restrict__ mf, float* __restrict__ denom) {
  int e = blockIdx.x * blockDim.x + threadIdx.x;
  if (e >= E + Nn) return;
  int s, d;
  if (e < E) { s = src[e]; d = dst[e]; } else { s = d = e - E; }
  float w = __expf(lrelu(as_[s] + ad_[d]) - mf[d]);
  __hip_atomic_fetch_add(&denom[d], w, __ATOMIC_RELAXED, __HIP_MEMORY_SCOPE_AGENT);
}

// one wave32 per edge: out[dst,:] += alpha * h[src,:]
__global__ void edge_scatter_kernel(const int* __restrict__ src, const int* __restrict__ dst,
                                    int E, int Nn,
                                    const float* __restrict__ as_, const float* __restrict__ ad_,
                                    const float* __restrict__ mf, const float* __restrict__ denom,
                                    const float* __restrict__ h, int HS,
                                    float* __restrict__ out, int OS, int H) {
  int e = (blockIdx.x * blockDim.x + threadIdx.x) >> 5;
  int lane = threadIdx.x & 31;
  if (e >= E + Nn) return;
  int s, d;
  if (e < E) { s = src[e]; d = dst[e]; } else { s = d = e - E; }
  float alpha = __expf(lrelu(as_[s] + ad_[d]) - mf[d]) / denom[d];
  const float* hrow = h + (size_t)s * HS;
  float* orow = out + (size_t)d * OS;
  for (int c = lane; c < H; c += 32)
    __hip_atomic_fetch_add(orow + c, alpha * hrow[c], __ATOMIC_RELAXED,
                           __HIP_MEMORY_SCOPE_AGENT);
}

// ---------------------------------------------------------------------------
extern "C" void kernel_launch(void* const* d_in, const int* in_sizes, int n_in,
                              void* d_out, int out_size, void* d_ws, size_t ws_size,
                              hipStream_t stream) {
  const float* x    = (const float*)d_in[0];
  const int*   ei   = (const int*)d_in[1];
  const float* W1   = (const float*)d_in[2];
  const float* aS1  = (const float*)d_in[3];
  const float* aD1  = (const float*)d_in[4];
  const float* b1   = (const float*)d_in[5];
  const float* W2s  = (const float*)d_in[6];
  const float* W2d  = (const float*)d_in[7];
  const float* aS2  = (const float*)d_in[8];
  const float* aD2  = (const float*)d_in[9];
  const float* b2   = (const float*)d_in[10];

  const int H  = in_sizes[3];            // 250
  const int F  = in_sizes[2] / H;        // 256
  const int Nn = in_sizes[0] / F;        // 50000
  const int E  = in_sizes[1] / 2;        // 800000
  const int Et = E + Nn;                 // + self loops
  const int Kp = (F + 31) & ~31;         // 256 (layer-1 K, padded)
  const int HS = (H + 31) & ~31;         // 256 (row stride / layer-2 K / padded N)
  const int* srcp = ei;
  const int* dstp = ei + E;

  // ---- workspace carve-up ----
  char* ws = (char*)d_ws;
  size_t off = 0;
  auto give = [&](size_t bytes) -> void* {
    void* p = ws + off;
    off = (off + bytes + 255) & ~(size_t)255;
    return p;
  };
  int bfCols = Kp > HS ? Kp : HS;
  unsigned short* xbf  = (unsigned short*)give((size_t)Nn * bfCols * 2); // x_bf16, later x1_bf16
  unsigned short* wt1  = (unsigned short*)give((size_t)HS * Kp * 2);
  unsigned short* wt2s = (unsigned short*)give((size_t)HS * HS * 2);
  unsigned short* wt2d = (unsigned short*)give((size_t)HS * HS * 2);
  float* hA    = (float*)give((size_t)Nn * HS * 4);   // h1, later h_s
  float* hB    = (float*)give((size_t)Nn * HS * 4);   // x1,  later h_d
  unsigned* mu = (unsigned*)give((size_t)Nn * 4);     // max keys, then decoded floats
  float* denom = (float*)give((size_t)Nn * 4);
  float* asb   = (float*)give((size_t)Nn * 4);
  float* adb   = (float*)give((size_t)Nn * 4);
  float* outf  = (float*)d_out;

  const int TB = 256;
  auto cdiv = [](long a, long b) -> unsigned { return (unsigned)((a + b - 1) / b); };
  dim3 gemmGrid((unsigned)((Nn + 15) / 16), (unsigned)((HS / 16 + 7) / 8));

  // ---- weights -> transposed bf16 (zero-padded) ----
  conv_w_kernel<<<cdiv((long)HS * Kp, TB), TB, 0, stream>>>(W1,  wt1,  F, H, Kp, HS);
  conv_w_kernel<<<cdiv((long)HS * HS, TB), TB, 0, stream>>>(W2s, wt2s, H, H, HS, HS);
  conv_w_kernel<<<cdiv((long)HS * HS, TB), TB, 0, stream>>>(W2d, wt2d, H, H, HS, HS);

  // ======================= LAYER 1 =======================
  cvt_bf16_kernel<<<cdiv((long)Nn * Kp, TB), TB, 0, stream>>>(x, F, F, xbf, Kp,
                                                              (long)Nn * Kp, 0);
  gemm_bf16_wmma<<<gemmGrid, TB, 0, stream>>>(xbf, wt1, hA, Nn, Kp, HS, HS);   // h1
  node_dots_kernel<<<cdiv((long)Nn * 32, TB), TB, 0, stream>>>(hA, aS1, hA, aD1,
                                                               asb, adb, Nn, H, HS);
  init_nodes_kernel<<<cdiv(Nn, TB), TB, 0, stream>>>(mu, denom, Nn);
  init_bias_kernel<<<cdiv((long)Nn * HS, TB), TB, 0, stream>>>(hB, b1, (long)Nn * HS, H, HS);
  edge_max_kernel<<<cdiv(Et, TB), TB, 0, stream>>>(srcp, dstp, E, Nn, asb, adb, mu);
  decode_m_kernel<<<cdiv(Nn, TB), TB, 0, stream>>>(mu, Nn);
  edge_denom_kernel<<<cdiv(Et, TB), TB, 0, stream>>>(srcp, dstp, E, Nn, asb, adb,
                                                     (const float*)mu, denom);
  edge_scatter_kernel<<<cdiv((long)Et * 32, TB), TB, 0, stream>>>(
      srcp, dstp, E, Nn, asb, adb, (const float*)mu, denom, hA, HS, hB, HS, H);

  // x1 = relu(out1) -> bf16 (K-padded)
  cvt_bf16_kernel<<<cdiv((long)Nn * HS, TB), TB, 0, stream>>>(hB, HS, H, xbf, HS,
                                                              (long)Nn * HS, 1);

  // ======================= LAYER 2 =======================
  gemm_bf16_wmma<<<gemmGrid, TB, 0, stream>>>(xbf, wt2s, hA, Nn, HS, HS, HS);  // h_s
  gemm_bf16_wmma<<<gemmGrid, TB, 0, stream>>>(xbf, wt2d, hB, Nn, HS, HS, HS);  // h_d
  node_dots_kernel<<<cdiv((long)Nn * 32, TB), TB, 0, stream>>>(hA, aS2, hB, aD2,
                                                               asb, adb, Nn, H, HS);
  init_nodes_kernel<<<cdiv(Nn, TB), TB, 0, stream>>>(mu, denom, Nn);
  init_bias_kernel<<<cdiv((long)Nn * H, TB), TB, 0, stream>>>(outf, b2, (long)Nn * H, H, H);
  edge_max_kernel<<<cdiv(Et, TB), TB, 0, stream>>>(srcp, dstp, E, Nn, asb, adb, mu);
  decode_m_kernel<<<cdiv(Nn, TB), TB, 0, stream>>>(mu, Nn);
  edge_denom_kernel<<<cdiv(Et, TB), TB, 0, stream>>>(srcp, dstp, E, Nn, asb, adb,
                                                     (const float*)mu, denom);
  edge_scatter_kernel<<<cdiv((long)Et * 32, TB), TB, 0, stream>>>(
      srcp, dstp, E, Nn, asb, adb, (const float*)mu, denom, hA, HS, outf, H, H);
}
